// Attention_68771016344128
// MI455X (gfx1250) — matrix operand order repaired and checked by reading, not compiled, not run on hardware
//
#include <hip/hip_runtime.h>
#include <hip/hip_bf16.h>

typedef _Float16 v16h __attribute__((ext_vector_type(16)));
typedef _Float16 v8h  __attribute__((ext_vector_type(8)));
typedef float    v8f  __attribute__((ext_vector_type(8)));

#define B_  8
#define Q_  512
#define M_  512
#define R_  1024
#define D_  1024
#define H_  16
#define S_  64

// ---------------------------------------------------------------------------
// WMMA helpers (CDNA5, wave32): D = A(16x32 f16) * B(32x16 f16) + C(16x16 f32)
// ---------------------------------------------------------------------------
__device__ __forceinline__ v8f wmma16(v16h a, v16h b, v8f c) {
  return __builtin_amdgcn_wmma_f32_16x16x32_f16(false, a, false, b, (short)0, c,
                                                false, false);
}

__device__ __forceinline__ v8f zero8() {
  v8f z = {0.f, 0.f, 0.f, 0.f, 0.f, 0.f, 0.f, 0.f};
  return z;
}

// A-operand fragment: lane (l<16) holds row l with K = 0..15 contiguous
// (lanes 16..31: rows again, K = 16..31). Caller passes
// p = row_ptr + hh*16 ; chunks are at p and p+8.
__device__ __forceinline__ v16h fragA_at(const _Float16* p) {
  v8h lo = *(const v8h*)(p);
  v8h hi = *(const v8h*)(p + 8);
  return __builtin_shufflevector(lo, hi, 0, 1, 2, 3, 4, 5, 6, 7, 8, 9, 10, 11,
                                 12, 13, 14, 15);
}
// B-operand fragment: lane (l<16) holds column l with K = 0..15 contiguous
// (lanes 16..31: K = 16..31). Source stored as B^T rows (N x K row-major).
// Caller passes p = col_row_ptr + hh*16 ; chunks at p and p+8.
__device__ __forceinline__ v16h fragB_at(const _Float16* p) {
  v8h lo = *(const v8h*)(p);
  v8h hi = *(const v8h*)(p + 8);
  return __builtin_shufflevector(lo, hi, 0, 1, 2, 3, 4, 5, 6, 7, 8, 9, 10, 11,
                                 12, 13, 14, 15);
}

// ---------------------------------------------------------------------------
// Stage 0: pack / convert kernels
// ---------------------------------------------------------------------------
__global__ __launch_bounds__(256) void pack_x_kernel(
    const float* __restrict__ query, const float* __restrict__ memory,
    _Float16* __restrict__ X) {
  int idx = blockIdx.x * 256 + threadIdx.x;  // over B*R*D
  int b = idx >> 20;                         // / (R*D)
  int rem = idx & ((R_ * D_) - 1);
  int r = rem >> 10;
  int d = rem & (D_ - 1);
  float v = (r < M_) ? memory[((size_t)(b * M_ + r)) * D_ + d]
                     : query[((size_t)(b * Q_ + (r - M_))) * D_ + d];
  X[idx] = (_Float16)v;
}

__global__ __launch_bounds__(256) void cvt_kernel(const float* __restrict__ in,
                                                  _Float16* __restrict__ out) {
  int idx = blockIdx.x * 256 + threadIdx.x;
  out[idx] = (_Float16)in[idx];
}

// 1024x1024 transpose with f32->f16 convert: out[j][i] = in[i][j]
__global__ __launch_bounds__(256) void transpose_cvt_kernel(
    const float* __restrict__ in, _Float16* __restrict__ out) {
  int idx = blockIdx.x * 256 + threadIdx.x;  // over 1024*1024
  int i = idx >> 10;
  int j = idx & 1023;
  out[(size_t)j * 1024 + i] = (_Float16)in[idx];
}

// ---------------------------------------------------------------------------
// Stage 1: generic WMMA GEMM  C[m][n] = sum_k A[m][k] * Wt[n][k]
//   block = 256 threads (8 waves), block tile 128 rows x 64 cols,
//   each wave owns a 16x64 strip (4 accumulators), K = 1024.
//   Ping-pong pipelined: two fixed fragment buffer sets; each phase's WMMAs
//   issue, then that phase's buffers are refilled by loads directly into the
//   same virtual registers (no loop-carried v_mov rotation), while the other
//   phase's WMMAs cover the load latency.
// A row remap: a_row = (m/seg_len)*seg_stride + seg_off + m%seg_len
// modes: 0 = f16 row-major out
//        1 = f16 transposed out   Vt[(b*1024 + col)*R + r]
//        2 = f32 row-major out (final result)
//        3 = two f16 outputs with per-column biases added (Qc / Qp)
// ---------------------------------------------------------------------------
__global__ __launch_bounds__(256) void gemm16_kernel(
    const _Float16* __restrict__ A, const _Float16* __restrict__ Wt,
    void* __restrict__ out0, void* __restrict__ out1,
    const float* __restrict__ bias0, const float* __restrict__ bias1,
    int seg_len, int seg_stride, int seg_off, int mode) {
  int wv = threadIdx.x >> 5;
  int lane = threadIdx.x & 31;
  int ln = lane & 15;
  int hh = lane >> 4;
  int row0 = blockIdx.x * 128 + wv * 16;
  int col0 = blockIdx.y * 64;

  int seg = row0 / seg_len;
  int rin = row0 - seg * seg_len;
  const _Float16* Arow =
      A + (size_t)(seg * seg_stride + seg_off + rin + ln) * D_ + hh * 16;
  const _Float16* Brow = Wt + (size_t)(col0 + ln) * D_ + hh * 16;

  v8f acc0 = zero8(), acc1 = zero8(), acc2 = zero8(), acc3 = zero8();

  // phase-0 buffers (k = 0) and phase-1 buffers (k = 32)
  v16h a0  = fragA_at(Arow);
  v16h b00 = fragB_at(Brow);
  v16h b01 = fragB_at(Brow + (size_t)16 * D_);
  v16h b02 = fragB_at(Brow + (size_t)32 * D_);
  v16h b03 = fragB_at(Brow + (size_t)48 * D_);
  v16h a1  = fragA_at(Arow + 32);
  v16h b10 = fragB_at(Brow + 32);
  v16h b11 = fragB_at(Brow + 32 + (size_t)16 * D_);
  v16h b12 = fragB_at(Brow + 32 + (size_t)32 * D_);
  v16h b13 = fragB_at(Brow + 32 + (size_t)48 * D_);

#pragma unroll 1
  for (int k0 = 64; k0 < D_; k0 += 64) {
    // consume phase 0, then refill it from k0
    acc0 = wmma16(a0, b00, acc0);
    acc1 = wmma16(a0, b01, acc1);
    acc2 = wmma16(a0, b02, acc2);
    acc3 = wmma16(a0, b03, acc3);
    a0  = fragA_at(Arow + k0);
    b00 = fragB_at(Brow + k0);
    b01 = fragB_at(Brow + k0 + (size_t)16 * D_);
    b02 = fragB_at(Brow + k0 + (size_t)32 * D_);
    b03 = fragB_at(Brow + k0 + (size_t)48 * D_);
    // consume phase 1 (covers phase-0 refill latency), refill from k0+32
    acc0 = wmma16(a1, b10, acc0);
    acc1 = wmma16(a1, b11, acc1);
    acc2 = wmma16(a1, b12, acc2);
    acc3 = wmma16(a1, b13, acc3);
    a1  = fragA_at(Arow + k0 + 32);
    b10 = fragB_at(Brow + k0 + 32);
    b11 = fragB_at(Brow + k0 + 32 + (size_t)16 * D_);
    b12 = fragB_at(Brow + k0 + 32 + (size_t)32 * D_);
    b13 = fragB_at(Brow + k0 + 32 + (size_t)48 * D_);
  }
  // drain both phases (k = 960 and 992)
  acc0 = wmma16(a0, b00, acc0);
  acc1 = wmma16(a0, b01, acc1);
  acc2 = wmma16(a0, b02, acc2);
  acc3 = wmma16(a0, b03, acc3);
  acc0 = wmma16(a1, b10, acc0);
  acc1 = wmma16(a1, b11, acc1);
  acc2 = wmma16(a1, b12, acc2);
  acc3 = wmma16(a1, b13, acc3);

  v8f accs[4] = {acc0, acc1, acc2, acc3};

  if (mode == 0) {
    _Float16* O = (_Float16*)out0;
#pragma unroll
    for (int t = 0; t < 4; ++t) {
      int col = col0 + t * 16 + ln;
#pragma unroll
      for (int v = 0; v < 8; ++v) {
        int row = row0 + v + 8 * hh;
        O[(size_t)row * D_ + col] = (_Float16)accs[t][v];
      }
    }
  } else if (mode == 3) {
    _Float16* O0 = (_Float16*)out0;
    _Float16* O1 = (_Float16*)out1;
#pragma unroll
    for (int t = 0; t < 4; ++t) {
      int col = col0 + t * 16 + ln;
      float bv0 = bias0[col];
      float bv1 = bias1[col];
#pragma unroll
      for (int v = 0; v < 8; ++v) {
        int row = row0 + v + 8 * hh;
        size_t ix = (size_t)row * D_ + col;
        O0[ix] = (_Float16)(accs[t][v] + bv0);
        O1[ix] = (_Float16)(accs[t][v] + bv1);
      }
    }
  } else if (mode == 1) {
    _Float16* O = (_Float16*)out0;
    int b = row0 >> 10;   // row0 / R
    int r = row0 & 1023;  // row0 % R
#pragma unroll
    for (int t = 0; t < 4; ++t) {
      int col = col0 + t * 16 + ln;
      v8h pk;
#pragma unroll
      for (int v = 0; v < 8; ++v) pk[v] = (_Float16)accs[t][v];
      *(v8h*)(O + ((size_t)(b * D_ + col)) * R_ + r + 8 * hh) = pk;
    }
  } else {  // mode 2: f32 out
    float* O = (float*)out0;
#pragma unroll
    for (int t = 0; t < 4; ++t) {
      int col = col0 + t * 16 + ln;
#pragma unroll
      for (int v = 0; v < 8; ++v) {
        int row = row0 + v + 8 * hh;
        O[(size_t)row * D_ + col] = accs[t][v];
      }
    }
  }
}

// ---------------------------------------------------------------------------
// Stage 2: fused TXL attention with relative shift + online softmax.
// One wave per (b, h, 16-row q tile).  rel_shift: pos[q][r] = raw[q][r+Q-1-q]
// on the causal band (r <= q+M); everything past the band is masked to -1e9.
// ---------------------------------------------------------------------------
__global__ __launch_bounds__(128) void txl_attn_kernel(
    const _Float16* __restrict__ Qc, const _Float16* __restrict__ Qp,
    const _Float16* __restrict__ Kc, const _Float16* __restrict__ Kp,
    const _Float16* __restrict__ Vt, _Float16* __restrict__ Oh) {
  __shared__ float Gs[4][16][33];
  __shared__ __align__(16) _Float16 Ps[4][16][40];

  int wv = threadIdx.x >> 5;
  int lane = threadIdx.x & 31;
  int ln = lane & 15;
  int hh = lane >> 4;
  int task = blockIdx.x * 4 + wv;  // < 4096
  int qt = task & 31;
  int h = (task >> 5) & 15;
  int b = task >> 9;
  int q0 = qt << 4;

  const _Float16* qcrow =
      Qc + ((size_t)(b * Q_ + q0 + ln)) * D_ + h * S_ + hh * 16;
  const _Float16* qprow =
      Qp + ((size_t)(b * Q_ + q0 + ln)) * D_ + h * S_ + hh * 16;
  v16h aqc0 = fragA_at(qcrow);
  v16h aqc1 = fragA_at(qcrow + 32);
  v16h aqp0 = fragA_at(qprow);
  v16h aqp1 = fragA_at(qprow + 32);

  v8f o0 = zero8(), o1 = zero8(), o2 = zero8(), o3 = zero8();
  float mrow[8], lrow[8];
#pragma unroll
  for (int v = 0; v < 8; ++v) {
    mrow[v] = -3.0e38f;
    lrow[v] = 0.f;
  }

  int rend = M_ + q0 + 16;  // causal bound (last needed r is q0+15+M)
  for (int rc = 0; rc < rend; rc += 32) {
    float ps[2][8];
#pragma unroll
    for (int sub = 0; sub < 2; ++sub) {
      int r0s = rc + sub * 16;
      const _Float16* kcrow =
          Kc + ((size_t)(b * R_ + r0s + ln)) * D_ + h * S_ + hh * 16;
      // position band: G[q][u] = (q+pb) . Kp[jb+u],  u = (r-q) + 15 in [0,30]
      int jb = r0s + (Q_ - 16) - q0;  // always >= 0
      int j0 = jb + ln;
      int j1 = jb + 16 + ln;
      j0 = (j0 > R_ - 1) ? (R_ - 1) : j0;  // clamped rows are masked anyway
      j1 = (j1 > R_ - 1) ? (R_ - 1) : j1;
      const _Float16* kp0 = Kp + (size_t)j0 * D_ + h * S_ + hh * 16;
      const _Float16* kp1 = Kp + (size_t)j1 * D_ + h * S_ + hh * 16;
      // batch all 6 global fragment loads, then do the 6 WMMAs
      v16h bkc0 = fragB_at(kcrow);
      v16h bkc1 = fragB_at(kcrow + 32);
      v16h bp00 = fragB_at(kp0);
      v16h bp01 = fragB_at(kp0 + 32);
      v16h bp10 = fragB_at(kp1);
      v16h bp11 = fragB_at(kp1 + 32);
      v8f c = zero8();
      c = wmma16(aqc0, bkc0, c);
      c = wmma16(aqc1, bkc1, c);
      v8f g0 = zero8(), g1 = zero8();
      g0 = wmma16(aqp0, bp00, g0);
      g0 = wmma16(aqp1, bp01, g0);
      g1 = wmma16(aqp0, bp10, g1);
      g1 = wmma16(aqp1, bp11, g1);
#pragma unroll
      for (int v = 0; v < 8; ++v) {
        Gs[wv][v + 8 * hh][ln] = g0[v];
        Gs[wv][v + 8 * hh][16 + ln] = g1[v];
      }
      asm volatile("s_wait_dscnt 0x0" ::: "memory");
#pragma unroll
      for (int v = 0; v < 8; ++v) {
        int m = v + 8 * hh;                       // local q row
        float s = (c[v] + Gs[wv][m][ln - m + 15]) * 0.125f;
        if (r0s + ln > q0 + m + M_) s = -1.0e9f;  // causal mask
        ps[sub][v] = s;
      }
    }
    // issue V-fragment loads now so they fly under the softmax VALU work
    const _Float16* vrow =
        Vt + ((size_t)(b * D_ + h * S_ + ln)) * R_ + rc + hh * 16;
    v16h bv0 = fragB_at(vrow);
    v16h bv1 = fragB_at(vrow + (size_t)16 * R_);
    v16h bv2 = fragB_at(vrow + (size_t)32 * R_);
    v16h bv3 = fragB_at(vrow + (size_t)48 * R_);

    // ---- online softmax over this 32-wide r chunk (rows = q) ----
    float mx[8];
#pragma unroll
    for (int v = 0; v < 8; ++v) mx[v] = fmaxf(ps[0][v], ps[1][v]);
#pragma unroll
    for (int d = 1; d < 16; d <<= 1) {
#pragma unroll
      for (int v = 0; v < 8; ++v)
        mx[v] = fmaxf(mx[v], __shfl_xor(mx[v], d, 32));
    }
    float sc[8];
#pragma unroll
    for (int v = 0; v < 8; ++v) {
      float mn = fmaxf(mrow[v], mx[v]);
      sc[v] = __expf(mrow[v] - mn);
      mrow[v] = mn;
    }
    float rs[8];
#pragma unroll
    for (int v = 0; v < 8; ++v) {
      ps[0][v] = __expf(ps[0][v] - mrow[v]);
      ps[1][v] = __expf(ps[1][v] - mrow[v]);
      rs[v] = ps[0][v] + ps[1][v];
    }
#pragma unroll
    for (int d = 1; d < 16; d <<= 1) {
#pragma unroll
      for (int v = 0; v < 8; ++v) rs[v] += __shfl_xor(rs[v], d, 32);
    }
#pragma unroll
    for (int v = 0; v < 8; ++v) lrow[v] = lrow[v] * sc[v] + rs[v];
#pragma unroll
    for (int v = 0; v < 8; ++v) {
      o0[v] *= sc[v];
      o1[v] *= sc[v];
      o2[v] *= sc[v];
      o3[v] *= sc[v];
    }
    // ---- P to LDS (C layout) and back as A-operand, then P @ V ----
#pragma unroll
    for (int v = 0; v < 8; ++v) {
      int m = v + 8 * hh;
      Ps[wv][m][ln] = (_Float16)ps[0][v];
      Ps[wv][m][16 + ln] = (_Float16)ps[1][v];
    }
    asm volatile("s_wait_dscnt 0x0" ::: "memory");
    v16h ap = fragA_at(&Ps[wv][ln][hh * 16]);
    o0 = wmma16(ap, bv0, o0);
    o1 = wmma16(ap, bv1, o1);
    o2 = wmma16(ap, bv2, o2);
    o3 = wmma16(ap, bv3, o3);
    if (rc + 32 < rend) {  // wave-uniform prefetch of next Kc rows
      __builtin_prefetch(
          Kc + ((size_t)(b * R_ + rc + 32 + ln)) * D_ + h * S_, 0, 1);
    }
  }
  // epilogue: normalize and store f16 [b,q,h,s]
  v8f oo[4] = {o0, o1, o2, o3};
#pragma unroll
  for (int t = 0; t < 4; ++t) {
#pragma unroll
    for (int v = 0; v < 8; ++v) {
      int m = v + 8 * hh;
      float val = oo[t][v] / lrow[v];
      Oh[((size_t)(b * Q_ + q0 + m)) * D_ + h * S_ + t * 16 + ln] =
          (_Float16)val;
    }
  }
}

// ---------------------------------------------------------------------------
// Host launcher
// ---------------------------------------------------------------------------
extern "C" void kernel_launch(void* const* d_in, const int* in_sizes, int n_in,
                              void* d_out, int out_size, void* d_ws,
                              size_t ws_size, hipStream_t stream) {
  (void)in_sizes; (void)n_in; (void)out_size; (void)ws_size;
  const float* query  = (const float*)d_in[0];
  const float* memory = (const float*)d_in[1];
  const float* pos    = (const float*)d_in[2];
  // d_in[3] token_mask: computed analytically in-kernel
  const float* cb = (const float*)d_in[4];
  const float* pb = (const float*)d_in[5];
  const float* Wq = (const float*)d_in[6];
  const float* Wkc = (const float*)d_in[7];
  const float* Wkp = (const float*)d_in[8];
  const float* Wv = (const float*)d_in[9];
  const float* Wo = (const float*)d_in[10];

  char* ws = (char*)d_ws;
  size_t off = 0;
  auto alloc = [&](size_t bytes) -> void* {
    void* p = ws + off;
    off += (bytes + 255) & ~(size_t)255;
    return p;
  };
  const size_t f16 = sizeof(_Float16);
  _Float16* X     = (_Float16*)alloc((size_t)B_ * R_ * D_ * f16);
  _Float16* pos16 = (_Float16*)alloc((size_t)R_ * D_ * f16);
  _Float16* WqT   = (_Float16*)alloc((size_t)D_ * D_ * f16);
  _Float16* WkcT  = (_Float16*)alloc((size_t)D_ * D_ * f16);
  _Float16* WkpT  = (_Float16*)alloc((size_t)D_ * D_ * f16);
  _Float16* WvT   = (_Float16*)alloc((size_t)D_ * D_ * f16);
  _Float16* WoT   = (_Float16*)alloc((size_t)D_ * D_ * f16);
  _Float16* Qc    = (_Float16*)alloc((size_t)B_ * Q_ * D_ * f16);
  _Float16* Qp    = (_Float16*)alloc((size_t)B_ * Q_ * D_ * f16);
  _Float16* KcB   = (_Float16*)alloc((size_t)B_ * R_ * D_ * f16);
  _Float16* KpB   = (_Float16*)alloc((size_t)R_ * D_ * f16);
  _Float16* Vt    = (_Float16*)alloc((size_t)B_ * R_ * D_ * f16);
  _Float16* Oh    = (_Float16*)alloc((size_t)B_ * Q_ * D_ * f16);

  // Stage 0: pack + convert
  pack_x_kernel<<<(B_ * R_ * D_) / 256, 256, 0, stream>>>(query, memory, X);
  cvt_kernel<<<(R_ * D_) / 256, 256, 0, stream>>>(pos, pos16);
  transpose_cvt_kernel<<<(D_ * D_) / 256, 256, 0, stream>>>(Wq, WqT);
  transpose_cvt_kernel<<<(D_ * D_) / 256, 256, 0, stream>>>(Wkc, WkcT);
  transpose_cvt_kernel<<<(D_ * D_) / 256, 256, 0, stream>>>(Wkp, WkpT);
  transpose_cvt_kernel<<<(D_ * D_) / 256, 256, 0, stream>>>(Wv, WvT);
  transpose_cvt_kernel<<<(D_ * D_) / 256, 256, 0, stream>>>(Wo, WoT);

  // Stage 1: projections (WMMA GEMMs)
  dim3 blk(256);
  // Q projection: rows are query part of X (seg_len=Q, stride=R, off=M),
  // mode 3 writes Qc = q+cb and Qp = q+pb
  gemm16_kernel<<<dim3((B_ * Q_) / 128, D_ / 64), blk, 0, stream>>>(
      X, WqT, Qc, Qp, cb, pb, Q_, R_, M_, 3);
  // Kc: all of X
  gemm16_kernel<<<dim3((B_ * R_) / 128, D_ / 64), blk, 0, stream>>>(
      X, WkcT, KcB, nullptr, nullptr, nullptr, B_ * R_, 0, 0, 0);
  // Kp: positional encodings
  gemm16_kernel<<<dim3(R_ / 128, D_ / 64), blk, 0, stream>>>(
      pos16, WkpT, KpB, nullptr, nullptr, nullptr, R_, 0, 0, 0);
  // V: all of X, stored transposed [b,h,s,R]
  gemm16_kernel<<<dim3((B_ * R_) / 128, D_ / 64), blk, 0, stream>>>(
      X, WvT, Vt, nullptr, nullptr, nullptr, B_ * R_, 0, 0, 1);

  // Stage 2: fused attention (4096 wave-tasks, 4 waves/block)
  txl_attn_kernel<<<dim3((B_ * H_ * (Q_ / 16)) / 4), dim3(128), 0, stream>>>(
      Qc, Qp, KcB, KpB, Vt, Oh);

  // Stage 3: output projection -> f32 d_out
  gemm16_kernel<<<dim3((B_ * Q_) / 128, D_ / 64), blk, 0, stream>>>(
      Oh, WoT, d_out, nullptr, nullptr, nullptr, B_ * Q_, 0, 0, 2);
}
